// FullyFrameAttention_1726576855779
// MI455X (gfx1250) — matrix operand order repaired and checked by reading, not compiled, not run on hardware
//
#include <hip/hip_runtime.h>

typedef __attribute__((ext_vector_type(16))) _Float16 v16h;
typedef __attribute__((ext_vector_type(8)))  _Float16 v8h;
typedef __attribute__((ext_vector_type(8)))  float    v8f;
typedef __attribute__((ext_vector_type(4)))  unsigned v4u;
typedef __attribute__((ext_vector_type(8)))  int      v8i;
typedef __attribute__((ext_vector_type(4)))  int      v4i;

#define HEADS 8
#define DH    40
#define DHP   64
#define SEQ   4096
#define CCH   320
#define KCH   64          // keys per chunk
#define NCH   (SEQ/KCH)   // 64 chunks
#define LPAD  72          // LDS row stride in halves (64 data + 4-dword TDM pad)

// Build a v16h A/B fragment from two contiguous 16-byte half-groups.
__device__ __forceinline__ v16h load2x8(const _Float16* p0, const _Float16* p1) {
    v8h lo = *(const v8h*)p0;
    v8h hi = *(const v8h*)p1;
    return __builtin_shufflevector(lo, hi, 0,1,2,3,4,5,6,7,8,9,10,11,12,13,14,15);
}

__device__ __forceinline__ v8f wmma16(v16h a, v16h b, v8f c) {
    return __builtin_amdgcn_wmma_f32_16x16x32_f16(false, a, false, b, (short)0, c, false, false);
}

// ---- TDM descriptor helpers (D# per CDNA5 ISA 8.3/8.4) ----
// 2D tile of 16-bit elements, LDS pad: +4 dwords every 32 dwords (row 64 halves -> stride 72).
__device__ __forceinline__ void tdm_load_2d_f16(unsigned lds_addr, const void* gaddr,
                                                unsigned tile0, unsigned tile1,
                                                unsigned tdim0, unsigned tdim1,
                                                unsigned stride0) {
    unsigned long long ga = (unsigned long long)gaddr;
    v4u g0;
    g0[0] = 1u;                                   // count=1, user descriptor
    g0[1] = lds_addr;                             // LDS byte address
    g0[2] = (unsigned)(ga & 0xffffffffu);         // global_addr[31:0]
    g0[3] = (unsigned)((ga >> 32) & 0x01ffffffu) | (2u << 30);  // addr[56:32] | type=2
    v8i g1;
    // data_size=1(2B) | pad_enable | pad_interval=4(32dw) | pad_amount=3(4dw)
    g1[0] = (int)(0x00010000u | (1u << 20) | (4u << 22) | (3u << 25));
    g1[1] = (int)(tdim0 << 16);                   // [47:32]=atomic_addr(0) | tensor_dim0 lo16
    g1[2] = (int)((tdim0 >> 16) | (tdim1 << 16)); // tensor_dim0 hi16 | tensor_dim1 lo16
    g1[3] = (int)((tdim1 >> 16) | (tile0 << 16)); // tensor_dim1 hi16 | tile_dim0
    g1[4] = (int)tile1;                           // tile_dim1 (0 => 1-D) | tile_dim2=0
    g1[5] = (int)stride0;                         // tensor_dim0_stride lo32 (elements)
    g1[6] = 0;                                    // stride0 hi16 | tensor_dim1_stride lo16
    g1[7] = 0;
    v4i z4 = {0, 0, 0, 0};
#if __clang_major__ >= 23
    v8i z8 = {0, 0, 0, 0, 0, 0, 0, 0};
    __builtin_amdgcn_tensor_load_to_lds(g0, g1, z4, z4, z8, 0);
#else
    __builtin_amdgcn_tensor_load_to_lds(g0, g1, z4, z4, 0);
#endif
}

// ---------------- conversion / init kernels ----------------

__global__ void cvt_f32_to_f16(const float* __restrict__ src, _Float16* __restrict__ dst, int n) {
    int i = blockIdx.x * 256 + threadIdx.x;
    if (i < n) dst[i] = (_Float16)src[i];
}

__global__ void cvt_w_transpose(const float* __restrict__ W, _Float16* __restrict__ WT) {
    int i = blockIdx.x * 256 + threadIdx.x;
    if (i < CCH * CCH) {
        int ci = i / CCH, co = i % CCH;
        WT[co * CCH + ci] = (_Float16)W[i];
    }
}

__global__ void zero_f16(_Float16* __restrict__ p, int n) {
    int i = blockIdx.x * 256 + threadIdx.x;
    if (i < n) p[i] = (_Float16)0.0f;
}

// ones-column trick: vT[h][40][s] = 1 -> P@V accumulator col 40 = softmax denominator
__global__ void ones_row_vT(_Float16* __restrict__ vT) {
    int i = blockIdx.x * 256 + threadIdx.x;
    if (i < HEADS * SEQ) {
        int h = i / SEQ, s = i % SEQ;
        vT[((size_t)h * DHP + DH) * SEQ + s] = (_Float16)1.0f;
    }
}

// ---------------- QKV projection (WMMA) ----------------
__global__ __launch_bounds__(256) void qkv_proj(
    const _Float16* __restrict__ hsf,
    const _Float16* __restrict__ WqT, const _Float16* __restrict__ WkT,
    const _Float16* __restrict__ WvT,
    _Float16* __restrict__ qf, _Float16* __restrict__ kf, _Float16* __restrict__ vT)
{
    int wid  = (blockIdx.x * 256 + threadIdx.x) >> 5;
    int lane = threadIdx.x & 31;
    int mat  = wid / 5120;
    int rem  = wid % 5120;
    int mt = rem / 20, nt = rem % 20;

    const _Float16* WT = (mat == 0) ? WqT : (mat == 1) ? WkT : WvT;

    int l15  = lane & 15;
    int ab0  = (lane < 16) ? 0 : 8;
    int bk0  = (lane < 16) ? 0 : 8;
    int arow = mt * 16 + l15;
    int bcol = nt * 16 + l15;

    v8f acc = {};
    for (int kb = 0; kb < CCH; kb += 32) {
        const _Float16* ap = hsf + (size_t)arow * CCH + kb;
        const _Float16* bp = WT  + (size_t)bcol * CCH + kb + bk0;
        acc = wmma16(load2x8(ap + ab0, ap + 16 + ab0), load2x8(bp, bp + 16), acc);
    }

    const float scale = 0.15811388300841897f;  // 40^-0.5, folded into Q
    int colc = nt * 16 + l15;
    int h = colc / DH, j = colc % DH;
    int rbase = mt * 16 + ((lane < 16) ? 0 : 8);
    #pragma unroll
    for (int r = 0; r < 8; ++r) {
        int row = rbase + r;
        float v = acc[r];
        if (mat == 0)      qf[((size_t)(h * SEQ) + row) * DHP + j] = (_Float16)(v * scale);
        else if (mat == 1) kf[((size_t)(h * SEQ) + row) * DHP + j] = (_Float16)v;
        else               vT[((size_t)(h * DHP) + j) * SEQ + row] = (_Float16)v;
    }
}

// ---------------- fused flash attention: TDM double-buffered K/V + WMMA ----------------
// 8 waves/block share one head; wave0 drives the Tensor Data Mover.
__global__ __launch_bounds__(256) void flash_attn(
    const _Float16* __restrict__ qf, const _Float16* __restrict__ kf,
    const _Float16* __restrict__ vT, _Float16* __restrict__ ao)
{
    __shared__ _Float16 Kl[2][64 * LPAD];   // staged K chunk  [key][dh], padded rows
    __shared__ _Float16 Vl[2][64 * LPAD];   // staged V chunk  [dh][key], padded rows
    __shared__ _Float16 Pl[8][16 * LPAD];   // per-wave P transpose slab

    int wslot = threadIdx.x >> 5;
    int lane  = threadIdx.x & 31;
    int tile  = blockIdx.x * 8 + wslot;     // 0..2047; all 8 share head = tile>>8
    int h  = tile >> 8;
    int mt = tile & 255;

    int l15 = lane & 15;
    int hi8 = (lane < 16) ? 0 : 8;
    int bk0 = (lane < 16) ? 0 : 8;

    const _Float16* qbase = qf + ((size_t)(h * SEQ) + mt * 16 + l15) * DHP;
    v16h qa0 = load2x8(qbase +      hi8, qbase + 16 + hi8);
    v16h qa1 = load2x8(qbase + 32 + hi8, qbase + 48 + hi8);

    v8f o[4] = {{}, {}, {}, {}};
    float m[8];
    #pragma unroll
    for (int r = 0; r < 8; ++r) m[r] = -1e30f;

    _Float16* Pbuf = &Pl[wslot][0];
    const _Float16* kg = kf + (size_t)h * SEQ * DHP;          // head K base (contiguous rows)
    const _Float16* vg = vT + (size_t)h * DHP * SEQ;          // head V base (row stride SEQ)

    // prologue: stage chunk 0 into buffer 0
    if (wslot == 0) {
        tdm_load_2d_f16((unsigned)(uintptr_t)&Kl[0][0], kg, 4096, 0, 4096, 1, 4096);
        tdm_load_2d_f16((unsigned)(uintptr_t)&Vl[0][0], vg, KCH, DHP, SEQ, DHP, SEQ);
    }

    for (int c = 0; c < NCH; ++c) {
        int buf = c & 1;
        if (wslot == 0) {
            if (c + 1 < NCH) {
                size_t kb = (size_t)(c + 1) * KCH;
                tdm_load_2d_f16((unsigned)(uintptr_t)&Kl[buf ^ 1][0], kg + kb * DHP,
                                4096, 0, 4096, 1, 4096);
                tdm_load_2d_f16((unsigned)(uintptr_t)&Vl[buf ^ 1][0], vg + kb,
                                KCH, DHP, SEQ, DHP, SEQ);
                __builtin_amdgcn_s_wait_tensorcnt(2);   // chunk c resident
            } else {
                __builtin_amdgcn_s_wait_tensorcnt(0);
            }
        }
        __syncthreads();                                // chunk c visible to all waves

        const _Float16* Kbuf = &Kl[buf][0];
        const _Float16* Vbuf = &Vl[buf][0];

        // --- scores: four 16x16 tiles over 64 keys ---
        v8f s[4];
        #pragma unroll
        for (int t = 0; t < 4; ++t) {
            const _Float16* kp = Kbuf + (16 * t + l15) * LPAD + bk0;
            v8f z = {};
            z = wmma16(qa0, load2x8(kp,      kp + 16),  z);
            s[t] = wmma16(qa1, load2x8(kp + 32, kp + 48), z);
        }

        // --- online softmax over 64 columns ---
        float nm[8];
        #pragma unroll
        for (int r = 0; r < 8; ++r) {
            float x = fmaxf(fmaxf(s[0][r], s[1][r]), fmaxf(s[2][r], s[3][r]));
            x = fmaxf(x, __shfl_xor(x, 1, 32));
            x = fmaxf(x, __shfl_xor(x, 2, 32));
            x = fmaxf(x, __shfl_xor(x, 4, 32));
            x = fmaxf(x, __shfl_xor(x, 8, 32));
            nm[r] = fmaxf(m[r], x);
        }
        #pragma unroll
        for (int r = 0; r < 8; ++r) {
            float cf = __expf(m[r] - nm[r]);
            m[r] = nm[r];
            o[0][r] *= cf; o[1][r] *= cf; o[2][r] *= cf; o[3][r] *= cf;
            _Float16* pr = Pbuf + (r + hi8) * LPAD;
            pr[l15]      = (_Float16)__expf(s[0][r] - nm[r]);
            pr[16 + l15] = (_Float16)__expf(s[1][r] - nm[r]);
            pr[32 + l15] = (_Float16)__expf(s[2][r] - nm[r]);
            pr[48 + l15] = (_Float16)__expf(s[3][r] - nm[r]);
        }
        asm volatile("s_wait_dscnt 0" ::: "memory");

        // --- P as two A-fragments (keys 0..31, 32..63) ---
        const _Float16* prow = Pbuf + l15 * LPAD;
        v16h pa0 = load2x8(prow +      hi8, prow + 16 + hi8);
        v16h pa1 = load2x8(prow + 32 + hi8, prow + 48 + hi8);

        // --- O += P @ V; denominator rides V's ones-column (dh=40) ---
        #pragma unroll
        for (int nt = 0; nt < 4; ++nt) {
            const _Float16* vp = Vbuf + (nt * 16 + l15) * LPAD;
            o[nt] = wmma16(pa0, load2x8(vp + bk0,      vp + bk0 + 16),  o[nt]);
            o[nt] = wmma16(pa1, load2x8(vp + 32 + bk0, vp + 48 + bk0), o[nt]);
        }
        __syncthreads();                                // buf reusable for TDM prefetch
    }

    // epilogue: l = o[2] column 40 (l15==8); divide & merge heads (f16)
    #pragma unroll
    for (int r = 0; r < 8; ++r) {
        float l   = __shfl(o[2][r], (lane < 16) ? 8 : 24, 32);
        float inv = 1.0f / l;
        int row = mt * 16 + r + hi8;
        size_t base = (size_t)row * CCH + h * DH;
        ao[base + l15]      = (_Float16)(o[0][r] * inv);
        ao[base + 16 + l15] = (_Float16)(o[1][r] * inv);
        if (l15 < 8) ao[base + 32 + l15] = (_Float16)(o[2][r] * inv);
    }
}

// ---------------- output projection + bias (WMMA) ----------------
__global__ __launch_bounds__(256) void out_proj(
    const _Float16* __restrict__ ao, const _Float16* __restrict__ WoT,
    const float* __restrict__ bo, float* __restrict__ out)
{
    int wid  = (blockIdx.x * 256 + threadIdx.x) >> 5;
    int lane = threadIdx.x & 31;
    int mt = wid / 20, nt = wid % 20;
    int l15  = lane & 15;
    int ab0  = (lane < 16) ? 0 : 8;
    int bk0  = (lane < 16) ? 0 : 8;
    int arow = mt * 16 + l15;
    int bcol = nt * 16 + l15;

    v8f acc = {};
    for (int kb = 0; kb < CCH; kb += 32) {
        const _Float16* ap = ao  + (size_t)arow * CCH + kb;
        const _Float16* bp = WoT + (size_t)bcol * CCH + kb + bk0;
        acc = wmma16(load2x8(ap + ab0, ap + 16 + ab0), load2x8(bp, bp + 16), acc);
    }
    int c = nt * 16 + l15;
    float bias = bo[c];
    int rbase = mt * 16 + ((lane < 16) ? 0 : 8);
    #pragma unroll
    for (int r = 0; r < 8; ++r) {
        out[(size_t)(rbase + r) * CCH + c] = acc[r] + bias;
    }
}

// ---------------- host launch ----------------

extern "C" void kernel_launch(void* const* d_in, const int* in_sizes, int n_in,
                              void* d_out, int out_size, void* d_ws, size_t ws_size,
                              hipStream_t stream) {
    const float* hs = (const float*)d_in[0];
    const float* Wq = (const float*)d_in[1];
    const float* Wk = (const float*)d_in[2];
    const float* Wv = (const float*)d_in[3];
    const float* Wo = (const float*)d_in[4];
    const float* bo = (const float*)d_in[5];
    float* out = (float*)d_out;

    char* ws = (char*)d_ws;
    const size_t HS_N   = (size_t)SEQ * CCH;
    const size_t W_N    = (size_t)CCH * CCH;
    const size_t HEAD_N = (size_t)HEADS * SEQ * DHP;

    _Float16* hsf = (_Float16*)ws;
    _Float16* WqT = (_Float16*)(ws + HS_N * 2);
    _Float16* WkT = WqT + W_N;
    _Float16* WvT = WkT + W_N;
    _Float16* WoT = WvT + W_N;
    _Float16* qf  = WoT + W_N;            // [h][S][64]
    _Float16* kf  = qf + HEAD_N;          // [h][S][64]
    _Float16* vT  = kf + HEAD_N;          // [h][64][S]
    _Float16* ao  = vT + HEAD_N;          // [S][320]

    cvt_f32_to_f16<<<(int)((HS_N + 255) / 256), 256, 0, stream>>>(hs, hsf, (int)HS_N);
    cvt_w_transpose<<<(int)((W_N + 255) / 256), 256, 0, stream>>>(Wq, WqT);
    cvt_w_transpose<<<(int)((W_N + 255) / 256), 256, 0, stream>>>(Wk, WkT);
    cvt_w_transpose<<<(int)((W_N + 255) / 256), 256, 0, stream>>>(Wv, WvT);
    cvt_w_transpose<<<(int)((W_N + 255) / 256), 256, 0, stream>>>(Wo, WoT);

    zero_f16<<<(int)((3 * HEAD_N + 255) / 256), 256, 0, stream>>>(qf, (int)(3 * HEAD_N));
    ones_row_vT<<<(HEADS * SEQ + 255) / 256, 256, 0, stream>>>(vT);

    qkv_proj<<<1920, 256, 0, stream>>>(hsf, WqT, WkT, WvT, qf, kf, vT);
    flash_attn<<<256, 256, 0, stream>>>(qf, kf, vT, ao);
    out_proj<<<640, 256, 0, stream>>>(ao, WoT, bo, out);
}
